// DotProductAttention_90855738179832
// MI455X (gfx1250) — matrix operand order repaired and checked by reading, not compiled, not run on hardware
//
#include <hip/hip_runtime.h>
#include <hip/hip_bf16.h>

// ---------------------------------------------------------------------------
// Fused attention for MI455X (gfx1250, wave32, WMMA 16x16x32 f16 + TDM).
//   qp = q@wq+bq (scaled), kp = k@wk+bk, vp = v@wv+bv
//   out = softmax(qp kp^T) vp      (flash-style, no SxS materialization)
// K/V tiles are staged into LDS by the Tensor Data Mover (tensor_load_to_lds),
// double-buffered, shared by all 4 waves of the block.
// ---------------------------------------------------------------------------

#define B_SZ   4
#define S_LEN  4096
#define DMODEL 1024
#define DKV    128

typedef __attribute__((ext_vector_type(16))) _Float16 v16h;
typedef __attribute__((ext_vector_type(8)))  _Float16 v8h;
typedef __attribute__((ext_vector_type(8)))  float    v8f;
typedef __attribute__((ext_vector_type(4)))  unsigned int u32x4;
typedef __attribute__((ext_vector_type(8)))  unsigned int u32x8;

union V16 { v16h v; v8h h[2]; };

__device__ __forceinline__ float rowmax16(float v) {
    v = fmaxf(v, __shfl_xor(v, 1, 32));
    v = fmaxf(v, __shfl_xor(v, 2, 32));
    v = fmaxf(v, __shfl_xor(v, 4, 32));
    v = fmaxf(v, __shfl_xor(v, 8, 32));
    return v;   // xor masks < 16 keep the reduction inside each 16-lane half
}
__device__ __forceinline__ float rowsum16(float v) {
    v += __shfl_xor(v, 1, 32);
    v += __shfl_xor(v, 2, 32);
    v += __shfl_xor(v, 4, 32);
    v += __shfl_xor(v, 8, 32);
    return v;
}

// LDS byte offset of a __shared__ object: flat aperture keeps the LDS offset
// in addr[31:0] (ISA 10.2 aperture mapping).
__device__ __forceinline__ unsigned lds_off(const void* p) {
    return (unsigned)(unsigned long long)p;
}

// ---------------------------------------------------------------------------
// TDM 2D tile load: rows x width halves (2B elems), row stride in elems,
// packed contiguously at LDS offset.  D# per ISA ch.8: group0 = count=1,
// lds_addr, global_addr[56:0], type=2; group1 = data_size=2B, tensor dims,
// tile dims, dim0 stride.  workgroup_mask=0 (not in a cluster).
// ---------------------------------------------------------------------------
__device__ __forceinline__ void tdm_load_2d(unsigned lds, const void* gptr,
                                            unsigned width, unsigned rows,
                                            unsigned stride) {
    unsigned long long ga = (unsigned long long)gptr;
    u32x4 g0;
    g0[0] = 1u;                                               // count=1, user mode
    g0[1] = lds;                                              // lds_addr
    g0[2] = (unsigned)ga;                                     // global_addr[31:0]
    g0[3] = (unsigned)((ga >> 32) & 0x01FFFFFFull) | (2u << 30); // ga[56:32], type=2
    u32x8 g1;
    g1[0] = (1u << 16);                                       // data_size=1 (2B)
    g1[1] = (width & 0xFFFFu) << 16;                          // tensor_dim0[15:0]
    g1[2] = (width >> 16) | ((rows & 0xFFFFu) << 16);         // dim0[31:16], dim1[15:0]
    g1[3] = (rows >> 16) | ((width & 0xFFFFu) << 16);         // dim1[31:16], tile_dim0
    g1[4] = rows & 0xFFFFu;                                   // tile_dim1 (tile_dim2=0)
    g1[5] = stride;                                           // tensor_dim0_stride[31:0]
    g1[6] = 0u;                                               // stride0[47:32], stride1 lo
    g1[7] = 0u;                                               // stride1 hi
    asm volatile("tensor_load_to_lds %0, %1" :: "s"(g0), "s"(g1) : "memory");
}

// ---------------------------------------------------------------------------
// Kernel 1: weights fp32 [1024][128] -> f16 transposed wT[3][128][1024]
// ---------------------------------------------------------------------------
__global__ void prep_weights(const float* __restrict__ wq,
                             const float* __restrict__ wk,
                             const float* __restrict__ wv,
                             _Float16* __restrict__ wT) {
    int idx = blockIdx.x * blockDim.x + threadIdx.x;
    if (idx >= DMODEL * DKV) return;
    int d = idx >> 7;
    int n = idx & (DKV - 1);
    size_t o = (size_t)n * DMODEL + d;
    wT[o]                        = (_Float16)wq[idx];
    wT[(size_t)DKV*DMODEL + o]   = (_Float16)wk[idx];
    wT[(size_t)2*DKV*DMODEL + o] = (_Float16)wv[idx];
}

// ---------------------------------------------------------------------------
// Kernel 2: projections via WMMA.  One wave = 16 rows x 128 cols.
// B fragments preloaded as a register array so the 16 b128 loads issue as a
// clause ahead of the 8 WMMAs (overlap instead of wait-0 per WMMA).
// ---------------------------------------------------------------------------
__global__ __launch_bounds__(128) void proj_kernel(
    const float* __restrict__ q, const float* __restrict__ k,
    const float* __restrict__ v,
    const float* __restrict__ bq, const float* __restrict__ bk,
    const float* __restrict__ bv,
    const _Float16* __restrict__ wT,
    _Float16* __restrict__ Qp, _Float16* __restrict__ Kp,
    _Float16* __restrict__ VpT) {
    const int lane = threadIdx.x & 31;
    const int wave = threadIdx.x >> 5;
    const int hf   = lane >> 4;
    const int l16  = lane & 15;
    const int proj = blockIdx.y;
    const int row0 = (blockIdx.x * 4 + wave) * 16;

    const float* x    = (proj == 0) ? q  : (proj == 1) ? k  : v;
    const float* bias = (proj == 0) ? bq : (proj == 1) ? bk : bv;
    const _Float16* w = wT + (size_t)proj * DKV * DMODEL;

    v8f acc[8];
#pragma unroll
    for (int t = 0; t < 8; ++t) acc[t] = v8f{};

    const float* xrow = x + (size_t)(row0 + l16) * DMODEL;  // A row = lane%16

    for (int kc = 0; kc < DMODEL; kc += 32) {
        // A fragment 16x32 f16: lanes0-15 K{0..15}, lanes16-31 K{16..31}
        V16 a;
        const float* p0 = xrow + kc + hf * 16;
        const float* p1 = xrow + kc + hf * 16 + 8;
#pragma unroll
        for (int i = 0; i < 8; ++i) {
            a.v[i]     = (_Float16)p0[i];
            a.v[8 + i] = (_Float16)p1[i];
        }
        v16h bfr[8];
#pragma unroll
        for (int t = 0; t < 8; ++t)
            bfr[t] = *(const v16h*)(w + (size_t)(t * 16 + l16) * DMODEL + kc + hf * 16);
#pragma unroll
        for (int t = 0; t < 8; ++t)
            acc[t] = __builtin_amdgcn_wmma_f32_16x16x32_f16(
                false, a.v, false, bfr[t], (short)0, acc[t], false, false);
    }

    const float scale = 0.088388347648318447f;  // 1/sqrt(128)
#pragma unroll
    for (int t = 0; t < 8; ++t) {
#pragma unroll
        for (int r = 0; r < 8; ++r) {
            int col  = t * 16 + l16;          // C: N = lane%16
            int grow = row0 + r + 8 * hf;     // C: M = r + 8*half
            float val = acc[t][r] + bias[col];
            if (proj == 0) {
                Qp[(size_t)grow * DKV + col] = (_Float16)(val * scale);
            } else if (proj == 1) {
                Kp[(size_t)grow * DKV + col] = (_Float16)val;
            } else {
                int b_ = grow >> 12;
                int s_ = grow & (S_LEN - 1);
                VpT[((size_t)(b_ * DKV + col)) * S_LEN + s_] = (_Float16)val;
            }
        }
    }
}

// ---------------------------------------------------------------------------
// Kernel 3: flash attention with TDM-staged, double-buffered K/V in LDS.
// One wave = one 16-query tile; 4 waves share each 32-key K/V block.
// Per block iteration: wave0 issues TDM loads for the next block, all waves
// compute S = Q K^T (8 WMMA) + online softmax + O += P V (8 WMMA) from the
// current LDS buffers, wave0 drains TENSORcnt, block barrier, swap buffers.
// ---------------------------------------------------------------------------
__global__ __launch_bounds__(128) void attn_kernel(
    const _Float16* __restrict__ Qp, const _Float16* __restrict__ Kp,
    const _Float16* __restrict__ VpT, float* __restrict__ out) {
    __shared__ _Float16 Kbuf[2][32][DKV];   // 16 KB: keys j..j+31, row-major d
    __shared__ _Float16 Vbuf[2][DKV][32];   // 16 KB: v-cols x keys j..j+31
    __shared__ _Float16 Pst[4][16][32];     // 4 KB : per-wave P staging tile

    const int lane  = threadIdx.x & 31;
    const int wave  = threadIdx.x >> 5;
    const int hf    = lane >> 4;
    const int l16   = lane & 15;
    const int batch = blockIdx.y;
    const int qrow0 = (blockIdx.x * 4 + wave) * 16;

    const _Float16* kg = Kp + (size_t)batch * S_LEN * DKV;
    const _Float16* vg = VpT + (size_t)batch * DKV * S_LEN;

    // Q fragments: 16 rows x 128 d as 4 A-frags, resident in VGPRs throughout
    V16 qa[4];
    {
        const _Float16* qb = Qp + (size_t)(batch * S_LEN + qrow0 + l16) * DKV;
#pragma unroll
        for (int c = 0; c < 4; ++c) {
            qa[c].h[0] = *(const v8h*)(qb + c * 32 + hf * 16);
            qa[c].h[1] = *(const v8h*)(qb + c * 32 + hf * 16 + 8);
        }
    }

    v8f o[8];
    float m[8], lsum[8];
#pragma unroll
    for (int t = 0; t < 8; ++t) o[t] = v8f{};
#pragma unroll
    for (int r = 0; r < 8; ++r) { m[r] = -1e30f; lsum[r] = 0.0f; }

    // Prologue: stage block 0 (TDM, per-wave op, EXEC ignored; wave 0 only)
    if (wave == 0) {
        tdm_load_2d(lds_off(&Kbuf[0][0][0]), kg, DKV, 32, DKV);
        tdm_load_2d(lds_off(&Vbuf[0][0][0]), vg, 32, DKV, S_LEN);
        __builtin_amdgcn_s_wait_tensorcnt(0);
    }
    __syncthreads();

    for (int j = 0; j < S_LEN; j += 32) {
        const int cur = (j >> 5) & 1;
        if (wave == 0 && j + 32 < S_LEN) {   // stage next block behind compute
            tdm_load_2d(lds_off(&Kbuf[cur ^ 1][0][0]),
                        kg + (size_t)(j + 32) * DKV, DKV, 32, DKV);
            tdm_load_2d(lds_off(&Vbuf[cur ^ 1][0][0]),
                        vg + (j + 32), 32, DKV, S_LEN);
        }

        // ---- S tiles from LDS: keys [j..j+15] -> c0, [j+16..j+31] -> c1
        v8f c0 = v8f{}, c1 = v8f{};
#pragma unroll
        for (int c = 0; c < 4; ++c) {
            v16h kb = *(const v16h*)(&Kbuf[cur][l16][c * 32 + hf * 16]);
            c0 = __builtin_amdgcn_wmma_f32_16x16x32_f16(
                false, qa[c].v, false, kb, (short)0, c0, false, false);
        }
#pragma unroll
        for (int c = 0; c < 4; ++c) {
            v16h kb = *(const v16h*)(&Kbuf[cur][16 + l16][c * 32 + hf * 16]);
            c1 = __builtin_amdgcn_wmma_f32_16x16x32_f16(
                false, qa[c].v, false, kb, (short)0, c1, false, false);
        }

        // ---- online softmax per row (row = r + 8*hf, replicated over a half)
#pragma unroll
        for (int r = 0; r < 8; ++r) {
            float tm    = rowmax16(fmaxf(c0[r], c1[r]));
            float mn    = fmaxf(m[r], tm);
            float alpha = __expf(m[r] - mn);
            float p0    = __expf(c0[r] - mn);
            float p1    = __expf(c1[r] - mn);
            lsum[r]     = lsum[r] * alpha + rowsum16(p0 + p1);
            m[r]        = mn;
            Pst[wave][r + 8 * hf][l16]      = (_Float16)p0;
            Pst[wave][r + 8 * hf][16 + l16] = (_Float16)p1;
#pragma unroll
            for (int t = 0; t < 8; ++t) o[t][r] *= alpha;
        }
        asm volatile("s_wait_dscnt 0" ::: "memory");  // same-wave P store->load
        // ---- P as A-fragment (16 rows x 32 keys)
        V16 pa;
        pa.h[0] = *(const v8h*)(&Pst[wave][l16][hf * 16]);
        pa.h[1] = *(const v8h*)(&Pst[wave][l16][hf * 16 + 8]);
        // ---- O += P * Vp from LDS (B-frag per 16-col tile)
#pragma unroll
        for (int t = 0; t < 8; ++t) {
            v16h vb = *(const v16h*)(&Vbuf[cur][t * 16 + l16][hf * 16]);
            o[t] = __builtin_amdgcn_wmma_f32_16x16x32_f16(
                false, pa.v, false, vb, (short)0, o[t], false, false);
        }

        if (wave == 0 && j + 32 < S_LEN)
            __builtin_amdgcn_s_wait_tensorcnt(0);  // next buffers resident
        __syncthreads();                           // publish to all waves
    }

    // ---- finalize: O /= l, write fp32 [B][S][128]
#pragma unroll
    for (int t = 0; t < 8; ++t) {
#pragma unroll
        for (int r = 0; r < 8; ++r) {
            out[((size_t)(batch * S_LEN + qrow0 + r + 8 * hf)) * DKV + t * 16 + l16] =
                o[t][r] / lsum[r];
        }
    }
}

// ---------------------------------------------------------------------------
extern "C" void kernel_launch(void* const* d_in, const int* in_sizes, int n_in,
                              void* d_out, int out_size, void* d_ws, size_t ws_size,
                              hipStream_t stream) {
    const float* q  = (const float*)d_in[0];
    const float* k  = (const float*)d_in[1];
    const float* v  = (const float*)d_in[2];
    const float* wq = (const float*)d_in[3];
    const float* bq = (const float*)d_in[4];
    const float* wk = (const float*)d_in[5];
    const float* bk = (const float*)d_in[6];
    const float* wv = (const float*)d_in[7];
    const float* bv = (const float*)d_in[8];
    float* out = (float*)d_out;

    char* ws = (char*)d_ws;
    // workspace layout (32B aligned):
    //   Qp  f16 [B*S][128]      : 4 MB
    //   Kp  f16 [B*S][128]      : 4 MB
    //   VpT f16 [B][128][S]     : 4 MB
    //   wT  f16 [3][128][1024]  : 768 KB
    _Float16* Qp  = (_Float16*)(ws);
    _Float16* Kp  = (_Float16*)(ws + (size_t)(4 << 20));
    _Float16* VpT = (_Float16*)(ws + (size_t)(8 << 20));
    _Float16* wT  = (_Float16*)(ws + (size_t)(12 << 20));

    prep_weights<<<(DMODEL * DKV + 255) / 256, 256, 0, stream>>>(wq, wk, wv, wT);
    proj_kernel<<<dim3((B_SZ * S_LEN) / (16 * 4), 3), 128, 0, stream>>>(
        q, k, v, bq, bk, bv, wT, Qp, Kp, VpT);
    attn_kernel<<<dim3(S_LEN / (16 * 4), B_SZ), 128, 0, stream>>>(Qp, Kp, VpT, out);
}
